// LinearAttention_62594853372349
// MI455X (gfx1250) — matrix operand order repaired and checked by reading, not compiled, not run on hardware
//
#include <hip/hip_runtime.h>
#include <hip/hip_bf16.h>
#include <math.h>

typedef __bf16 bf16_t;
typedef __bf16 v8bf  __attribute__((ext_vector_type(8)));
typedef __bf16 v16bf __attribute__((ext_vector_type(16)));
typedef float  v8f   __attribute__((ext_vector_type(8)));

#define D_MODEL  1024
#define NHEADS   16
#define DHEAD    64
#define SEQ      4096
#define BATCH    4
#define NMEM     4
#define QK_SCALE 0.125f   // 64^-0.5

// ---------------------------------------------------------------------------
// Pre-pass A: x f32 -> bf16 (row-major 16384 x 1024). One-time conversion so
// the GEMM's 24 column-block passes over the A panels stream bf16 from L2
// (32MB resident in the 192MB L2) with no VALU conversion in the hot loop.
// ---------------------------------------------------------------------------
__global__ __launch_bounds__(256) void conv_x(
    const float* __restrict__ x, bf16_t* __restrict__ xb)
{
  size_t seg = (size_t)blockIdx.x * 256 + threadIdx.x;   // 8 elems per seg
  const float* src = x + seg * 8;
  float4 f0 = *(const float4*)(src);
  float4 f1 = *(const float4*)(src + 4);
  v8bf o;
  o[0] = (bf16_t)f0.x; o[1] = (bf16_t)f0.y; o[2] = (bf16_t)f0.z; o[3] = (bf16_t)f0.w;
  o[4] = (bf16_t)f1.x; o[5] = (bf16_t)f1.y; o[6] = (bf16_t)f1.z; o[7] = (bf16_t)f1.w;
  *(v8bf*)(xb + seg * 8) = o;
}

// ---------------------------------------------------------------------------
// Pre-pass B: w_qkv f32 -> bf16 with the row permutation folded in:
// output row j' = s*1024 + h*64 + d  <-  source row j = h*192 + d*3 + s
// (this linearizes the reference's reshape(..., h, d_head, 3) interleave so
// the GEMM epilogue can scatter straight into Q/K/V buffers).
// ---------------------------------------------------------------------------
__global__ __launch_bounds__(256) void conv_w(
    const float* __restrict__ w, bf16_t* __restrict__ wb)
{
  int seg = blockIdx.x * 256 + threadIdx.x;   // 3072*1024/8 = 393216 segs
  int jp = seg >> 7, c = (seg & 127) * 8;
  int sel = jp >> 10, rem = jp & 1023;
  int h = rem >> 6, d = rem & 63;
  int j = h * 192 + d * 3 + sel;
  const float* src = w + (size_t)j * D_MODEL + c;
  float4 f0 = *(const float4*)(src);
  float4 f1 = *(const float4*)(src + 4);
  v8bf o;
  o[0] = (bf16_t)f0.x; o[1] = (bf16_t)f0.y; o[2] = (bf16_t)f0.z; o[3] = (bf16_t)f0.w;
  o[4] = (bf16_t)f1.x; o[5] = (bf16_t)f1.y; o[6] = (bf16_t)f1.z; o[7] = (bf16_t)f1.w;
  *(v8bf*)(wb + (size_t)jp * D_MODEL + c) = o;
}

// ---------------------------------------------------------------------------
// Kernel 1: QKV projection GEMM, bf16 WMMA + async-to-LDS double buffering.
//   C(16384 x 3072) = xb(16384 x 1024) @ wb^T
// Block tile 128x128, 8 waves (2x4), wave tile 64x32, BK=64, 2 LDS buffers.
// Staging uses GLOBAL_LOAD_ASYNC_TO_LDS_B128 (ASYNCcnt-tracked, bypasses
// VGPRs); each thread issues 8 b128 loads per tile, waits with
// s_wait_asynccnt while the next tile is still in flight (<=8 outstanding).
// ---------------------------------------------------------------------------
__global__ __launch_bounds__(256) void qkv_gemm_wmma(
    const bf16_t* __restrict__ xb, const bf16_t* __restrict__ wb,
    float* __restrict__ Qb, float* __restrict__ Kb, float* __restrict__ Vb)
{
  // row stride 72 elems = 144B = 9*16B -> every b128 dest is 16B aligned
  __shared__ __align__(16) bf16_t As[2][128][72];
  __shared__ __align__(16) bf16_t Bs[2][128][72];

  const int tid  = threadIdx.x;
  const int lane = tid & 31, wid = tid >> 5;
  const int wm = wid >> 2, wn = wid & 3;       // 2 x 4 waves
  const int bm = blockIdx.x * 128;
  const int bn = blockIdx.y * 128;
  const int half = lane >> 4, l16 = lane & 15;

  v8f zero = {0.f,0.f,0.f,0.f,0.f,0.f,0.f,0.f};
  v8f acc[4][2];
#pragma unroll
  for (int i = 0; i < 4; ++i)
#pragma unroll
    for (int j = 0; j < 2; ++j) acc[i][j] = zero;

  // issue one 128x64 bf16 tile of A and B into LDS buffer `buf`
  auto issue_tile = [&](int step, int buf) {
    const int k0 = step * 64;
#pragma unroll
    for (int i = 0; i < 4; ++i) {
      int s = tid + 256 * i;              // 1024 16B segs per matrix
      int r = s >> 3, c = (s & 7) * 8;
      // LDS byte offset: flat shared address truncates to LDS offset (ISA 10.2)
      uint32_t aoff = (uint32_t)(uintptr_t)&As[buf][r][c];
      uint32_t boff = (uint32_t)(uintptr_t)&Bs[buf][r][c];
      uint32_t av = (uint32_t)(((bm + r) * D_MODEL + k0 + c) * 2);
      uint32_t bv = (uint32_t)(((bn + r) * D_MODEL + k0 + c) * 2);
      asm volatile("global_load_async_to_lds_b128 %0, %1, %2"
                   :: "v"(aoff), "v"(av), "s"(xb) : "memory");
      asm volatile("global_load_async_to_lds_b128 %0, %1, %2"
                   :: "v"(boff), "v"(bv), "s"(wb) : "memory");
    }
  };

  constexpr int NSTEP = D_MODEL / 64;     // 16
  issue_tile(0, 0);

  for (int step = 0; step < NSTEP; ++step) {
    const int buf = step & 1;
    if (step + 1 < NSTEP) {
      issue_tile(step + 1, buf ^ 1);      // prefetch next tile
      asm volatile("s_wait_asynccnt 0x8" ::: "memory");  // current tile landed
    } else {
      asm volatile("s_wait_asynccnt 0x0" ::: "memory");
    }
    __syncthreads();                      // all threads' async writes visible

#pragma unroll
    for (int ksub = 0; ksub < 2; ++ksub) {
      const int kb = ksub * 32;
      // B fragments (32x16): lane%16 = col, lane/16 selects K half
      v16bf bfrag[2];
#pragma unroll
      for (int nt = 0; nt < 2; ++nt) {
        int n = wn * 32 + nt * 16 + l16;
        v8bf b0 = *(const v8bf*)&Bs[buf][n][kb + half * 16];
        v8bf b1 = *(const v8bf*)&Bs[buf][n][kb + half * 16 + 8];
        bfrag[nt] = __builtin_shufflevector(b0, b1,
            0,1,2,3,4,5,6,7,8,9,10,11,12,13,14,15);
      }
      // A fragments (16x32) + WMMA
#pragma unroll
      for (int mt = 0; mt < 4; ++mt) {
        int r = wm * 64 + mt * 16 + l16;
        v8bf a0 = *(const v8bf*)&As[buf][r][kb + half * 16];      // K a..a+7
        v8bf a1 = *(const v8bf*)&As[buf][r][kb + half * 16 + 8];  // K a+8..a+15
        v16bf af = __builtin_shufflevector(a0, a1,
            0,1,2,3,4,5,6,7,8,9,10,11,12,13,14,15);
#pragma unroll
        for (int nt = 0; nt < 2; ++nt)
          acc[mt][nt] = __builtin_amdgcn_wmma_f32_16x16x32_bf16(
              false, af, false, bfrag[nt], (short)0, acc[mt][nt], false, false);
      }
    }
    __syncthreads();                      // reads done before buffer reuse
  }

  // epilogue: C layout = VGPR g: lanes0-15 M=g, lanes16-31 M=g+8; N=l16
#pragma unroll
  for (int nt = 0; nt < 2; ++nt) {
    int jp = bn + wn * 32 + nt * 16 + l16;
    int sel = jp >> 10, rem = jp & 1023;
    int h = rem >> 6, d = rem & 63;
    float* buf = (sel == 0) ? Qb : ((sel == 1) ? Kb : Vb);
#pragma unroll
    for (int mt = 0; mt < 4; ++mt) {
      int rbase = bm + wm * 64 + mt * 16 + half * 8;
#pragma unroll
      for (int g = 0; g < 8; ++g) {
        int r = rbase + g;
        int b_ = r >> 12, l_ = r & 4095;
        buf[(((size_t)(b_ * NHEADS + h) * SEQ) + l_) * DHEAD + d] = acc[mt][nt][g];
      }
    }
  }
}

// ---------------------------------------------------------------------------
// Kernel 2: per-(b,h,d) column max and sum-exp of K over {mem rows + unmasked
// sequence rows}. Masked rows are excluded (== -inf in the reference).
// ---------------------------------------------------------------------------
__global__ __launch_bounds__(256) void kstats(
    const float* __restrict__ Kb, const int* __restrict__ mask,
    const float* __restrict__ memkv,
    float* __restrict__ colmax, float* __restrict__ colsum)
{
  int bh = blockIdx.x, b = bh >> 4, h = bh & 15;
  int t = threadIdx.x, d = t & 63, q = t >> 6;
  const float* Kp = Kb + (size_t)bh * SEQ * DHEAD;
  const int* mp = mask + b * SEQ;

  float mx = -INFINITY;
  for (int l = q * 1024; l < q * 1024 + 1024; ++l)
    if (mp[l]) mx = fmaxf(mx, Kp[(size_t)l * DHEAD + d]);
  if (q == 0)
    for (int n = 0; n < NMEM; ++n)
      mx = fmaxf(mx, memkv[h * 256 + n * 64 + d]);

  __shared__ float red[4][64];
  red[q][d] = mx;
  __syncthreads();
  mx = fmaxf(fmaxf(red[0][d], red[1][d]), fmaxf(red[2][d], red[3][d]));
  __syncthreads();

  float sm = 0.f;
  for (int l = q * 1024; l < q * 1024 + 1024; ++l)
    if (mp[l]) sm += __expf(Kp[(size_t)l * DHEAD + d] - mx);
  if (q == 0)
    for (int n = 0; n < NMEM; ++n)
      sm += __expf(memkv[h * 256 + n * 64 + d] - mx);
  red[q][d] = sm;
  __syncthreads();
  if (q == 0) {
    colmax[bh * 64 + d] = mx;
    colsum[bh * 64 + d] = red[0][d] + red[1][d] + red[2][d] + red[3][d];
  }
}

// ---------------------------------------------------------------------------
// Kernel 3: partial context ctx[d][e] += exp(K[n,d]-max[d]) * V[n,e] over an
// 8-way split of the sequence; chunk 0 also folds in the 4 memory kv rows.
// ---------------------------------------------------------------------------
__global__ __launch_bounds__(256) void ctx_partial(
    const float* __restrict__ Kb, const float* __restrict__ Vb,
    const int* __restrict__ mask, const float* __restrict__ memkv,
    const float* __restrict__ colmax, float* __restrict__ ctx_part)
{
  int bh = blockIdx.x, chunk = blockIdx.y;
  int b = bh >> 4, h = bh & 15;
  int t = threadIdx.x;
  int d = t >> 2, eb = (t & 3) * 16;

  __shared__ float cmax[64];
  __shared__ float Kw[16][64];
  __shared__ float Vv[16][64];
  if (t < 64) cmax[t] = colmax[bh * 64 + t];
  __syncthreads();

  const float* Kp = Kb + (size_t)bh * SEQ * DHEAD;
  const float* Vp = Vb + (size_t)bh * SEQ * DHEAD;
  const int* mp = mask + b * SEQ;

  float c[16];
#pragma unroll
  for (int i = 0; i < 16; ++i) c[i] = 0.f;

  int l0 = chunk * 512;
  for (int stage = 0; stage < 32; ++stage) {
    int base = l0 + stage * 16;
#pragma unroll
    for (int i = 0; i < 4; ++i) {
      int idx = t + 256 * i;
      int r = idx >> 6, cc = idx & 63;
      int l = base + r;
      float kv = Kp[(size_t)l * 64 + cc];
      Kw[r][cc] = mp[l] ? __expf(kv - cmax[cc]) : 0.f;
      Vv[r][cc] = Vp[(size_t)l * 64 + cc];
    }
    __syncthreads();
#pragma unroll
    for (int r = 0; r < 16; ++r) {
      float wv = Kw[r][d];
#pragma unroll
      for (int e = 0; e < 16; ++e) c[e] = fmaf(wv, Vv[r][eb + e], c[e]);
    }
    __syncthreads();
  }

  if (chunk == 0) {
    for (int n = 0; n < NMEM; ++n) {
      float wv = __expf(memkv[h * 256 + n * 64 + d] - cmax[d]);
      const float* mvp = memkv + NHEADS * NMEM * DHEAD + h * 256 + n * 64;
#pragma unroll
      for (int e = 0; e < 16; ++e) c[e] = fmaf(wv, mvp[eb + e], c[e]);
    }
  }

  float* outp = ctx_part + ((size_t)(bh * 8 + chunk) * 64 + d) * 64 + eb;
#pragma unroll
  for (int e = 0; e < 16; ++e) outp[e] = c[e];
}

// ---------------------------------------------------------------------------
// Kernel 4: reduce the 8 partial contexts and normalize by sum-exp.
// ---------------------------------------------------------------------------
__global__ __launch_bounds__(256) void ctx_reduce(
    const float* __restrict__ ctx_part, const float* __restrict__ colsum,
    float* __restrict__ ctx)
{
  int bh = blockIdx.x, t = threadIdx.x;
  for (int idx = t; idx < 4096; idx += 256) {
    int d = idx >> 6;
    float s = 0.f;
#pragma unroll
    for (int p = 0; p < 8; ++p)
      s += ctx_part[(size_t)(bh * 8 + p) * 4096 + idx];
    ctx[(size_t)bh * 4096 + idx] = s / colsum[bh * 64 + d];
  }
}

// ---------------------------------------------------------------------------
// Kernel 5: per-row q softmax (over d=64) + 64x64 matvec with context.
// One wave32 per head; 8 waves/block, gridDim.y=2 covers all 16 heads.
// Mask check is block-uniform, so the early-out never splits a barrier.
// ---------------------------------------------------------------------------
__global__ __launch_bounds__(256) void attn_out(
    const float* __restrict__ Qb, const float* __restrict__ ctx,
    const int* __restrict__ mask, float* __restrict__ out)
{
  int row = blockIdx.x;                   // b*4096 + l
  int b = row >> 12, l = row & 4095;
  int wid = threadIdx.x >> 5, lane = threadIdx.x & 31;
  int h = blockIdx.y * 8 + wid;
  int bh = b * NHEADS + h;
  float* op = out + (size_t)row * D_MODEL + h * DHEAD;

  if (mask[row] == 0) {                   // uniform across block
    op[lane] = 0.f;
    op[lane + 32] = 0.f;
    return;
  }

  const float* qp = Qb + ((size_t)bh * SEQ + l) * DHEAD;
  float q0 = qp[lane] * QK_SCALE;
  float q1 = qp[lane + 32] * QK_SCALE;
  float mx = fmaxf(q0, q1);
#pragma unroll
  for (int off = 16; off > 0; off >>= 1)
    mx = fmaxf(mx, __shfl_xor(mx, off, 32));
  float e0 = __expf(q0 - mx), e1 = __expf(q1 - mx);
  float s = e0 + e1;
#pragma unroll
  for (int off = 16; off > 0; off >>= 1)
    s += __shfl_xor(s, off, 32);
  float inv = 1.f / s;

  __shared__ float prob[8][64];
  prob[wid][lane] = e0 * inv;
  prob[wid][lane + 32] = e1 * inv;
  __syncthreads();

  const float* cp = ctx + (size_t)bh * 4096;   // 16KB/head, L2-resident
  float a0 = 0.f, a1 = 0.f;
#pragma unroll 4
  for (int d = 0; d < 64; ++d) {
    float p = prob[wid][d];
    a0 = fmaf(p, cp[d * 64 + lane], a0);
    a1 = fmaf(p, cp[d * 64 + lane + 32], a1);
  }
  op[lane] = a0;
  op[lane + 32] = a1;
}

// ---------------------------------------------------------------------------
extern "C" void kernel_launch(void* const* d_in, const int* in_sizes, int n_in,
                              void* d_out, int out_size, void* d_ws, size_t ws_size,
                              hipStream_t stream)
{
  const float* x     = (const float*)d_in[0];   // (4, 4096, 1024)
  const float* w     = (const float*)d_in[1];   // (3072, 1024)
  const float* memkv = (const float*)d_in[2];   // (2, 16, 4, 64)
  const int*   mask  = (const int*)d_in[3];     // (4, 4096)
  float* out = (float*)d_out;                   // (4, 4096, 1024)

  char* ws = (char*)d_ws;
  size_t off = 0;
  const size_t bhl_d = (size_t)BATCH * NHEADS * SEQ * DHEAD;   // 16M elems
  float* Qb = (float*)(ws + off); off += bhl_d * sizeof(float);
  float* Kb = (float*)(ws + off); off += bhl_d * sizeof(float);
  float* Vb = (float*)(ws + off); off += bhl_d * sizeof(float);
  bf16_t* xb = (bf16_t*)(ws + off); off += bhl_d * sizeof(bf16_t);
  bf16_t* wb = (bf16_t*)(ws + off); off += (size_t)3 * D_MODEL * D_MODEL * sizeof(bf16_t);
  float* colmax = (float*)(ws + off); off += (size_t)64 * 64 * sizeof(float);
  float* colsum = (float*)(ws + off); off += (size_t)64 * 64 * sizeof(float);
  float* ctx_part = (float*)(ws + off); off += (size_t)64 * 8 * 4096 * sizeof(float);
  float* ctx = (float*)(ws + off); off += (size_t)64 * 4096 * sizeof(float);
  (void)ws_size; (void)in_sizes; (void)n_in; (void)out_size;

  conv_x<<<(BATCH * SEQ * D_MODEL / 8) / 256, 256, 0, stream>>>(x, xb);
  conv_w<<<(3 * D_MODEL * D_MODEL / 8) / 256, 256, 0, stream>>>(w, wb);

  dim3 gGemm((BATCH * SEQ) / 128, (3 * D_MODEL) / 128);        // 128 x 24
  qkv_gemm_wmma<<<gGemm, 256, 0, stream>>>(xb, wb, Qb, Kb, Vb);
  kstats<<<BATCH * NHEADS, 256, 0, stream>>>(Kb, mask, memkv, colmax, colsum);
  ctx_partial<<<dim3(BATCH * NHEADS, 8), 256, 0, stream>>>(Kb, Vb, mask, memkv,
                                                           colmax, ctx_part);
  ctx_reduce<<<BATCH * NHEADS, 256, 0, stream>>>(ctx_part, colsum, ctx);
  attn_out<<<dim3(BATCH * SEQ, 2), 256, 0, stream>>>(Qb, ctx, mask, out);
}
